// PointerNet_77309411328478
// MI455X (gfx1250) — matrix operand order repaired and checked by reading, not compiled, not run on hardware
//
#include <hip/hip_runtime.h>

typedef __attribute__((ext_vector_type(16))) _Float16 v16h;
typedef __attribute__((ext_vector_type(8)))  _Float16 v8h;
typedef __attribute__((ext_vector_type(8)))  float    v8f;

#define S_LEN 512
#define B_N   128
#define E_DIM 256
#define H_DIM 256
#define G_DIM 1024  // 4*H
#define BP    264   // padded LDS row stride in halfs: 132 words == 4 mod 64 banks

__device__ __forceinline__ float sigf(float x) { return 1.0f / (1.0f + expf(-x)); }

// Assemble A-fragment (16x32 f16, M x K) for one lane.
// Lane layout (this file's contiguous k order, not ISA table order): m = lane&15, half = lane>>4.
// elements 0..7  (VGPR0..3): K in [16*half, 16*half+8)
// elements 8..15 (VGPR4..7): K in [16*half+8, 16*half+16)
__device__ __forceinline__ v16h load_a_frag(const _Float16* __restrict__ rowp,
                                            int k0, int half) {
  v8h lo = *(const v8h*)(rowp + k0 + 16 * half);
  v8h hi = *(const v8h*)(rowp + k0 + 16 * half + 8);
  v16h a;
#pragma unroll
  for (int i = 0; i < 8; ++i) { a[i] = lo[i]; a[i + 8] = hi[i]; }
  return a;
}

// B-fragment (32x16 f16, K x N) from an LDS row: col n = lane&15,
// K in [k0 + 16*half, k0 + 16*half + 16) -> 32 contiguous bytes (16B aligned).
__device__ __forceinline__ v16h load_b_frag_lds(const _Float16* rowp, int k0,
                                                int half) {
  v8h lo = *(const v8h*)(rowp + k0 + 16 * half);
  v8h hi = *(const v8h*)(rowp + k0 + 16 * half + 8);
  v16h b;
#pragma unroll
  for (int i = 0; i < 8; ++i) { b[i] = lo[i]; b[i + 8] = hi[i]; }
  return b;
}

// ---------------------------------------------------------------------------
// gates[128,1024] = di16[128,256] @ Wih16[1024,256]^T
//                 + h16[128,256]  @ Whh16[1024,256]^T + b_ih + b_hh
// One wave per 16x16 D-tile: wave = M-tile (8), blockIdx.x = N-tile (64).
// The two 16-row B panels (shared by all 8 waves) are staged in LDS via
// CDNA5 async global->LDS copies (ASYNCcnt), then WMMA reads B from LDS.
// ---------------------------------------------------------------------------
__global__ void k_gates(const _Float16* __restrict__ di16,
                        const _Float16* __restrict__ h16,
                        const _Float16* __restrict__ Wih,
                        const _Float16* __restrict__ Whh,
                        const float* __restrict__ b_ih,
                        const float* __restrict__ b_hh,
                        float* __restrict__ gates) {
  __shared__ _Float16 shB[2][16 * BP];  // [matrix][n_local * BP + k]

  const int wave = threadIdx.x >> 5;
  const int lane = threadIdx.x & 31;
  const int half = lane >> 4;
  const int lr   = lane & 15;
  const int tile_m = wave;         // 128 / 16 = 8 tiles
  const int tile_n = blockIdx.x;   // 1024 / 16 = 64 tiles
  const int m  = tile_m * 16 + lr;
  const int n0 = tile_n * 16;
  const int n  = n0 + lr;

  // --- async-stage B panels: 32 rows (16 per matrix), 4 rows per wave.
  // Each lane copies 16B; one wave-issue moves one full 512B row.
#pragma unroll
  for (int r = 0; r < 4; ++r) {
    const int rowid = wave * 4 + r;      // 0..31
    const int mat   = rowid >> 4;        // 0: Wih, 1: Whh
    const int nl    = rowid & 15;
    const _Float16* gsrc =
        (mat ? Whh : Wih) + (size_t)(n0 + nl) * E_DIM + lane * 8;
    unsigned ldsa =
        (unsigned)(size_t)(&shB[mat][nl * BP]) + (unsigned)(lane * 16);
    asm volatile("global_load_async_to_lds_b128 %0, %1, off"
                 :
                 : "v"(ldsa), "v"(gsrc)
                 : "memory");
  }
  asm volatile("s_wait_asynccnt 0x0" ::: "memory");
  __syncthreads();

  const _Float16* arow_d = di16 + (size_t)m * E_DIM;
  const _Float16* arow_h = h16  + (size_t)m * H_DIM;
  const _Float16* brow_i = &shB[0][lr * BP];
  const _Float16* brow_h = &shB[1][lr * BP];

  v8f acc = {};
#pragma unroll
  for (int k0 = 0; k0 < E_DIM; k0 += 32) {
    v16h a  = load_a_frag(arow_d, k0, half);
    v16h bm = load_b_frag_lds(brow_i, k0, half);
    acc = __builtin_amdgcn_wmma_f32_16x16x32_f16(false, a, false, bm,
                                                 (short)0, acc, false, false);
  }
#pragma unroll
  for (int k0 = 0; k0 < H_DIM; k0 += 32) {
    v16h a  = load_a_frag(arow_h, k0, half);
    v16h bm = load_b_frag_lds(brow_h, k0, half);
    acc = __builtin_amdgcn_wmma_f32_16x16x32_f16(false, a, false, bm,
                                                 (short)0, acc, false, false);
  }
  const float bias = b_ih[n] + b_hh[n];
#pragma unroll
  for (int r = 0; r < 8; ++r) {
    const int row = tile_m * 16 + r + 8 * half;
    gates[(size_t)row * G_DIM + n] = acc[r] + bias;
  }
}

// ---------------------------------------------------------------------------
// encT[n][s] = (x[512,256] @ Wa_enc[256,256]^T)[s][n] + b_a[n]   (transposed!)
// One-time kernel; plain global loads are fine here.
// ---------------------------------------------------------------------------
__global__ void k_encproj(const _Float16* __restrict__ x16,
                          const _Float16* __restrict__ Wenc16,
                          const float* __restrict__ b_a,
                          float* __restrict__ encT) {
  const int wave = threadIdx.x >> 5;
  const int lane = threadIdx.x & 31;
  const int half = lane >> 4;
  const int lr   = lane & 15;
  const int tile   = blockIdx.x * 8 + wave;  // 512 tiles
  const int tile_m = tile >> 4;              // 32 tiles over S=512
  const int tile_n = tile & 15;              // 16 tiles over H=256
  const int m = tile_m * 16 + lr;
  const int n = tile_n * 16 + lr;

  const _Float16* arow = x16    + (size_t)m * E_DIM;
  const _Float16* brow = Wenc16 + (size_t)n * E_DIM;

  v8f acc = {};
#pragma unroll
  for (int k0 = 0; k0 < E_DIM; k0 += 32) {
    v16h a  = load_a_frag(arow, k0, half);
    v16h bm = *(const v16h*)(brow + k0 + 16 * half);
    acc = __builtin_amdgcn_wmma_f32_16x16x32_f16(false, a, false, bm,
                                                 (short)0, acc, false, false);
  }
  const float bias = b_a[n];
#pragma unroll
  for (int r = 0; r < 8; ++r) {
    const int s = tile_m * 16 + r + 8 * half;
    encT[(size_t)n * S_LEN + s] = acc[r] + bias;  // stored [H, S]
  }
}

// ---------------------------------------------------------------------------
// Fused per-step kernel. One block per batch row b (512 threads, 16 waves):
//   LSTM pointwise -> dec_proj matvec (coalesced via transposed Wa_dec) ->
//   attention logits -> masked log-softmax -> Gumbel argmax -> updates.
// Block b exclusively owns c[b], mask[b], lps[b], di[b] -> no races.
// ---------------------------------------------------------------------------
__global__ void k_step(const float* __restrict__ gates,
                       float* __restrict__ cbuf,
                       _Float16* __restrict__ h16,
                       _Float16* __restrict__ di16,
                       const _Float16* __restrict__ x16,
                       const float* __restrict__ encT,
                       const float* __restrict__ vvec,
                       const float* __restrict__ WadT,  // [k][j] transposed
                       int* __restrict__ maskb,
                       float* __restrict__ lps,
                       const float* __restrict__ gumbel,
                       int t,
                       float* __restrict__ out) {
  __shared__ float sh_h[H_DIM];
  __shared__ float sh_dp[H_DIM];
  __shared__ float sh_v[H_DIM];
  __shared__ float sh_logit[S_LEN];
  __shared__ float sh_r[S_LEN];
  __shared__ int   sh_i[S_LEN];

  const int b = blockIdx.x;
  const int tid = threadIdx.x;

  // Phase 1: LSTM cell (torch gate order i,f,g,o)
  if (tid < H_DIM) {
    const float ig = gates[(size_t)b * G_DIM + tid];
    const float fg = gates[(size_t)b * G_DIM + H_DIM + tid];
    const float gg = gates[(size_t)b * G_DIM + 2 * H_DIM + tid];
    const float og = gates[(size_t)b * G_DIM + 3 * H_DIM + tid];
    const float cc = cbuf[(size_t)b * H_DIM + tid];
    const float cn = sigf(fg) * cc + sigf(ig) * tanhf(gg);
    const float hh = sigf(og) * tanhf(cn);
    cbuf[(size_t)b * H_DIM + tid] = cn;
    sh_h[tid] = hh;
    h16[(size_t)b * H_DIM + tid] = (_Float16)hh;
    sh_v[tid] = vvec[tid];
  }
  __syncthreads();

  // Phase 2: dec_proj[j] = sum_k h[b,k] * WadT[k][j]  (coalesced in j)
  if (tid < H_DIM) {
    float acc = 0.0f;
    for (int k = 0; k < H_DIM; ++k)
      acc = fmaf(sh_h[k], WadT[(size_t)k * H_DIM + tid], acc);
    sh_dp[tid] = acc;
  }
  __syncthreads();

  // Phase 3: attention logit for s = tid (encT is [H,S] -> coalesced)
  const float gval = gumbel[((size_t)t * B_N + b) * S_LEN + tid];
  {
    float acc = 0.0f;
    const float* ep = encT + tid;
    for (int k = 0; k < H_DIM; ++k)
      acc = fmaf(sh_v[k], tanhf(sh_dp[k] + ep[(size_t)k * S_LEN]), acc);
    const float lg = 10.0f * tanhf(acc);
    const bool mk = maskb[(size_t)b * S_LEN + tid] != 0;
    sh_logit[tid] = mk ? lg : -__builtin_inff();
  }
  __syncthreads();

  // max reduce
  sh_r[tid] = sh_logit[tid];
  __syncthreads();
  for (int st = S_LEN / 2; st > 0; st >>= 1) {
    if (tid < st) sh_r[tid] = fmaxf(sh_r[tid], sh_r[tid + st]);
    __syncthreads();
  }
  const float mx = sh_r[0];
  __syncthreads();

  // sum(exp) reduce  (masked: exp(-inf)=0)
  sh_r[tid] = expf(sh_logit[tid] - mx);
  __syncthreads();
  for (int st = S_LEN / 2; st > 0; st >>= 1) {
    if (tid < st) sh_r[tid] += sh_r[tid + st];
    __syncthreads();
  }
  const float lse = mx + logf(sh_r[0]);
  __syncthreads();

  // argmax(logit + gumbel) — first-index tiebreak like jnp.argmax
  sh_r[tid] = sh_logit[tid] + gval;
  sh_i[tid] = tid;
  __syncthreads();
  for (int st = S_LEN / 2; st > 0; st >>= 1) {
    if (tid < st) {
      const float o = sh_r[tid + st];
      const int  io = sh_i[tid + st];
      if (o > sh_r[tid] || (o == sh_r[tid] && io < sh_i[tid])) {
        sh_r[tid] = o; sh_i[tid] = io;
      }
    }
    __syncthreads();
  }
  const int sel = sh_i[0];

  if (tid == 0) {
    const float nl = lps[b] + (sh_logit[sel] - lse);
    lps[b] = nl;
    out[(size_t)b * S_LEN + t] = (float)sel;     // predicted_mappings[b][t]
    out[(size_t)B_N * S_LEN + b] = nl;           // log_probs_sum[b]
    maskb[(size_t)b * S_LEN + sel] = 0;
  }
  if (tid < E_DIM) di16[(size_t)b * E_DIM + tid] = x16[(size_t)sel * E_DIM + tid];
}

__global__ void k_conv(const float* __restrict__ src, _Float16* __restrict__ dst,
                       int n) {
  const int i = blockIdx.x * blockDim.x + threadIdx.x;
  if (i < n) dst[i] = (_Float16)src[i];
}

// One-time transpose: WadT[k*256 + j] = Wa_dec[j*256 + k]
__global__ void k_transpose(const float* __restrict__ src,
                            float* __restrict__ dst) {
  const int idx = blockIdx.x * blockDim.x + threadIdx.x;  // 0..65535
  const int k = idx >> 8;
  const int j = idx & 255;
  dst[(size_t)k * H_DIM + j] = src[(size_t)j * H_DIM + k];
}

__global__ void k_init(float* __restrict__ cbuf, _Float16* __restrict__ h16,
                       _Float16* __restrict__ di16, int* __restrict__ maskb,
                       float* __restrict__ lps,
                       const float* __restrict__ init_dec) {
  const int idx = blockIdx.x * blockDim.x + threadIdx.x;  // 0..65535
  if (idx < B_N * S_LEN) maskb[idx] = 1;
  if (idx < B_N * H_DIM) { cbuf[idx] = 0.0f; h16[idx] = (_Float16)0.0f; }
  if (idx < B_N * E_DIM) di16[idx] = (_Float16)init_dec[idx & (E_DIM - 1)];
  if (idx < B_N) lps[idx] = 0.0f;
}

extern "C" void kernel_launch(void* const* d_in, const int* in_sizes, int n_in,
                              void* d_out, int out_size, void* d_ws,
                              size_t ws_size, hipStream_t stream) {
  const float* x        = (const float*)d_in[0];
  const float* gumbel   = (const float*)d_in[1];
  const float* W_ih     = (const float*)d_in[2];
  const float* W_hh     = (const float*)d_in[3];
  const float* b_ih     = (const float*)d_in[4];
  const float* b_hh     = (const float*)d_in[5];
  const float* Wa_dec   = (const float*)d_in[6];
  const float* Wa_enc   = (const float*)d_in[7];
  const float* b_a      = (const float*)d_in[8];
  const float* vvec     = (const float*)d_in[9];
  const float* init_dec = (const float*)d_in[10];

  size_t off = 0;
  auto carve = [&](size_t bytes) -> void* {
    void* r = (char*)d_ws + off;
    off += (bytes + 255) & ~(size_t)255;
    return r;
  };
  float*    gates  = (float*)carve((size_t)B_N * G_DIM * sizeof(float));
  float*    cbuf   = (float*)carve((size_t)B_N * H_DIM * sizeof(float));
  _Float16* h16    = (_Float16*)carve((size_t)B_N * H_DIM * sizeof(_Float16));
  _Float16* di16   = (_Float16*)carve((size_t)B_N * E_DIM * sizeof(_Float16));
  float*    encT   = (float*)carve((size_t)H_DIM * S_LEN * sizeof(float));
  float*    WadT   = (float*)carve((size_t)H_DIM * H_DIM * sizeof(float));
  int*      maskb  = (int*)carve((size_t)B_N * S_LEN * sizeof(int));
  float*    lps    = (float*)carve((size_t)B_N * sizeof(float));
  _Float16* Wih16  = (_Float16*)carve((size_t)G_DIM * E_DIM * sizeof(_Float16));
  _Float16* Whh16  = (_Float16*)carve((size_t)G_DIM * H_DIM * sizeof(_Float16));
  _Float16* Wenc16 = (_Float16*)carve((size_t)H_DIM * E_DIM * sizeof(_Float16));
  _Float16* x16    = (_Float16*)carve((size_t)S_LEN * E_DIM * sizeof(_Float16));

  // One-time precompute (re-done every call: deterministic, capture-safe)
  k_conv<<<(G_DIM * E_DIM + 255) / 256, 256, 0, stream>>>(W_ih, Wih16, G_DIM * E_DIM);
  k_conv<<<(G_DIM * H_DIM + 255) / 256, 256, 0, stream>>>(W_hh, Whh16, G_DIM * H_DIM);
  k_conv<<<(H_DIM * E_DIM + 255) / 256, 256, 0, stream>>>(Wa_enc, Wenc16, H_DIM * E_DIM);
  k_conv<<<(S_LEN * E_DIM + 255) / 256, 256, 0, stream>>>(x, x16, S_LEN * E_DIM);
  k_transpose<<<(H_DIM * H_DIM + 255) / 256, 256, 0, stream>>>(Wa_dec, WadT);
  k_encproj<<<64, 256, 0, stream>>>(x16, Wenc16, b_a, encT);
  k_init<<<(B_N * S_LEN + 255) / 256, 256, 0, stream>>>(cbuf, h16, di16, maskb,
                                                        lps, init_dec);
  // Sequential decode: 2 kernels per step
  for (int t = 0; t < S_LEN; ++t) {
    k_gates<<<64, 256, 0, stream>>>(di16, h16, Wih16, Whh16, b_ih, b_hh, gates);
    k_step<<<B_N, S_LEN, 0, stream>>>(gates, cbuf, h16, di16, x16, encT, vvec,
                                      WadT, maskb, lps, gumbel, t,
                                      (float*)d_out);
  }
  (void)in_sizes; (void)n_in; (void)out_size; (void)ws_size;
}
